// QwemmaAttention_33243046871660
// MI455X (gfx1250) — matrix operand order repaired and checked by reading, not compiled, not run on hardware
//
#include <hip/hip_runtime.h>
#include <hip/hip_bf16.h>

// ---------------- problem constants ----------------
#define B_    2
#define T_    2048
#define D_    2048
#define NH_   16
#define NKV_  8
#define HD_   128
#define SCALAR_ 0.08838834764831845f
#define KMASK_  -2.3819763e38f

typedef __attribute__((ext_vector_type(16))) __bf16 v16bf;
typedef __attribute__((ext_vector_type(8)))  float  v8f;

union FragU { v16bf v; uint4 q[2]; };

// ---- CDNA5 async global->LDS (ISA 10 / 15.18.3, tracked by ASYNCcnt) ----
// INST_OFFSET is added to BOTH the LDS destination and the global source, so
// contiguous 16B chunks need only one base address pair + offset: immediates.
static __device__ inline unsigned lds_off(const void* p) {
  // generic pointer to __shared__: low 32 bits == LDS byte offset (ISA 10.2)
  return (unsigned)(unsigned long long)p;
}
#define ASYNC_LD(ldsaddr, gaddr, OFS)                                      \
  asm volatile("global_load_async_to_lds_b128 %0, %1, off offset:" #OFS    \
               :: "v"(ldsaddr), "v"(gaddr) : "memory")
static __device__ inline void wait_async0() {
  asm volatile("s_wait_asynccnt 0x0" ::: "memory");
}

// A-operand fragment (16x32 bf16): lane<16 -> row=lane, K [0..7] and [16..23];
// lane>=16 -> row=lane-16, K [8..15] and [24..31]   (ISA 7.12.2, 16-bit A layout)
static __device__ inline v16bf load_frag_a(const __bf16* base, int stride, int lane) {
  const int row = lane & 15;
  const int klo = (lane >> 4) << 3;
  const __bf16* p = base + (size_t)row * stride + klo;
  FragU u;
  u.q[0] = *(const uint4*)(p);
  u.q[1] = *(const uint4*)(p + 16);
  return u.v;
}

// B-operand fragment (32x16 bf16) loaded from Bt rows (N x K):
// lane<16 -> col N=lane, K [0..7] and [16..23]; lane>=16 -> col N=lane-16, K [8..15] and [24..31]
static __device__ inline v16bf load_frag_b(const __bf16* base, int stride, int lane) {
  const int row = lane & 15;
  const int k0  = (lane >> 4) << 3;
  const __bf16* p = base + (size_t)row * stride + k0;
  FragU u;
  u.q[0] = *(const uint4*)(p);
  u.q[1] = *(const uint4*)(p + 16);
  return u.v;
}

static __device__ inline v8f wmma_bf16(v16bf a, v16bf b, v8f c) {
  return __builtin_amdgcn_wmma_f32_16x16x32_bf16(false, a, false, b, (short)0, c, false, false);
}

// ---------------- pack kernels (fp32 -> bf16, transposed weight layouts) ----------------
__global__ void pack_x_kernel(const float* __restrict__ x, __bf16* __restrict__ xb) {
  int i = blockIdx.x * 256 + threadIdx.x;
  xb[i] = (__bf16)x[i];
}

// q_w (N,D,H) -> wq_t[(n*H+h)*D + d]
__global__ void pack_wq_kernel(const float* __restrict__ qw, __bf16* __restrict__ wq) {
  int i = blockIdx.x * 256 + threadIdx.x;          // 2048*2048
  int out = i >> 11, d = i & 2047;
  int n = out >> 7, h = out & 127;
  wq[i] = (__bf16)qw[((size_t)n * D_ + d) * HD_ + h];
}

// kv_w (2,KV,D,H) -> wk_t / wv_t [(kv*H+h)*D + d]
__global__ void pack_wkv_kernel(const float* __restrict__ kvw,
                                __bf16* __restrict__ wk, __bf16* __restrict__ wv) {
  int i = blockIdx.x * 256 + threadIdx.x;          // 2 * 1024*2048
  int c = i >> 21;
  int r = i & ((1 << 21) - 1);
  int out = r >> 11, d = r & 2047;
  int kvi = out >> 7, h = out & 127;
  float val = kvw[(((size_t)c * NKV_ + kvi) * D_ + d) * HD_ + h];
  (c ? wv : wk)[r] = (__bf16)val;
}

// o_w (NH, D) -> wo_t[dcol*NH + k]  (transpose)
__global__ void pack_wo_kernel(const float* __restrict__ ow, __bf16* __restrict__ wo) {
  int i = blockIdx.x * 256 + threadIdx.x;          // 2048*2048
  int ncol = i >> 11, kk = i & 2047;
  wo[i] = (__bf16)ow[(size_t)kk * D_ + ncol];
}

// ---------------- tiled WMMA GEMM:  C(MxN) = A(MxK) x Bt(NxK)^T ----------------
// Double-buffered LDS tiles fed by global_load_async_to_lds_b128.
// EPI 0: fp32 row-major out.  EPI 2: bf16 out scattered to vt[(b,kv,h,s)].
template <int EPI>
__global__ __launch_bounds__(256) void gemm_kernel(const __bf16* __restrict__ A,
                                                   const __bf16* __restrict__ Bt,
                                                   void* __restrict__ Cout,
                                                   int M, int N, int K) {
  __shared__ __bf16 As[2][128][40];
  __shared__ __bf16 Bs[2][128][40];

  const int tid  = threadIdx.x;
  const int lane = tid & 31;
  const int wave = tid >> 5;
  const int wm   = wave >> 2;          // 0..1  -> 64 rows
  const int wn   = wave & 3;           // 0..3  -> 32 cols
  const int m0   = blockIdx.y * 128;
  const int n0   = blockIdx.x * 128;

  v8f acc[4][2];
#pragma unroll
  for (int i = 0; i < 4; ++i)
#pragma unroll
    for (int j = 0; j < 2; ++j)
#pragma unroll
      for (int v = 0; v < 8; ++v) acc[i][j][v] = 0.f;

  const int ldr = tid >> 1;
  const int ldc = (tid & 1) << 4;
  const __bf16* pa_base = A  + (size_t)(m0 + ldr) * K + ldc;
  const __bf16* pb_base = Bt + (size_t)(n0 + ldr) * K + ldc;
  const unsigned la0 = lds_off(&As[0][ldr][ldc]);
  const unsigned lb0 = lds_off(&Bs[0][ldr][ldc]);
  const unsigned la1 = lds_off(&As[1][ldr][ldc]);
  const unsigned lb1 = lds_off(&Bs[1][ldr][ldc]);

  // prologue: async-fill buffer 0 with tile k0=0 (offset covers both sides)
  ASYNC_LD(la0, pa_base, 0);
  ASYNC_LD(la0, pa_base, 16);
  ASYNC_LD(lb0, pb_base, 0);
  ASYNC_LD(lb0, pb_base, 16);

  int buf = 0;
  for (int k0 = 0; k0 < K; k0 += 32) {
    wait_async0();        // this wave's async fills of `buf` are done
    __syncthreads();      // everyone's fills done; everyone's reads of buf^1 done
    if (k0 + 32 < K) {    // async-fill the other buffer with the next tile
      const unsigned la = buf ? la0 : la1;
      const unsigned lb = buf ? lb0 : lb1;
      const __bf16* pa = pa_base + (k0 + 32);
      const __bf16* pb = pb_base + (k0 + 32);
      ASYNC_LD(la, pa, 0);
      ASYNC_LD(la, pa, 16);
      ASYNC_LD(lb, pb, 0);
      ASYNC_LD(lb, pb, 16);
      if (k0 + 64 < K) {  // prefetch one more tile ahead into L2
        __builtin_prefetch(pa_base + (k0 + 64), 0, 1);
        __builtin_prefetch(pb_base + (k0 + 64), 0, 1);
      }
    }

    v16bf af[4], bfr[2];
#pragma unroll
    for (int mt = 0; mt < 4; ++mt) af[mt] = load_frag_a(&As[buf][wm * 64 + mt * 16][0], 40, lane);
#pragma unroll
    for (int nt = 0; nt < 2; ++nt) bfr[nt] = load_frag_b(&Bs[buf][wn * 32 + nt * 16][0], 40, lane);
#pragma unroll
    for (int mt = 0; mt < 4; ++mt)
#pragma unroll
      for (int nt = 0; nt < 2; ++nt)
        acc[mt][nt] = wmma_bf16(af[mt], bfr[nt], acc[mt][nt]);

    buf ^= 1;
  }

  const int rb = (lane >> 4) << 3;   // C layout: rows 0-7 in lanes 0-15, rows 8-15 in lanes 16-31
#pragma unroll
  for (int mt = 0; mt < 4; ++mt)
#pragma unroll
    for (int nt = 0; nt < 2; ++nt) {
      const int col  = n0 + wn * 32 + nt * 16 + (lane & 15);
      const int rowb = m0 + wm * 64 + mt * 16 + rb;
#pragma unroll
      for (int v = 0; v < 8; ++v) {
        const int row = rowb + v;
        const float val = acc[mt][nt][v];
        if (EPI == 0) {
          ((float*)Cout)[(size_t)row * N + col] = val;
        } else {  // EPI == 2 : V written transposed -> vt[((b*KV+kv)*H+h)*T + s]
          const int b = row >> 11, s = row & (T_ - 1);
          const int kv = col >> 7, h = col & 127;
          ((__bf16*)Cout)[(((size_t)b * NKV_ + kv) * HD_ + h) * T_ + s] = (__bf16)val;
        }
      }
    }
}

// ---------------- RMS-norm + RoPE (one wave per (token, head)) ----------------
__global__ __launch_bounds__(32) void normrope_kernel(const float* __restrict__ qf,
                                                      const float* __restrict__ kf,
                                                      const int* __restrict__ segpos,
                                                      const float* __restrict__ qscale,
                                                      const float* __restrict__ kscale,
                                                      __bf16* __restrict__ qb,
                                                      __bf16* __restrict__ kb) {
  const int bid   = blockIdx.x;
  const int token = bid / 24;
  const int j     = bid - token * 24;     // 0..15 = q heads, 16..23 = k heads
  const int lane  = threadIdx.x;
  const int pos   = segpos[token];

  const float* src; const float* scale; __bf16* dst;
  if (j < NH_) {
    src = qf + (size_t)token * (NH_ * HD_) + j * HD_;
    scale = qscale;
    dst = qb + (size_t)token * (NH_ * HD_) + j * HD_;
  } else {
    const int kvh = j - NH_;
    src = kf + (size_t)token * (NKV_ * HD_) + kvh * HD_;
    scale = kscale;
    dst = kb + (size_t)token * (NKV_ * HD_) + kvh * HD_;
  }

  float4 xv = *(const float4*)(src + lane * 4);
  float ss = xv.x * xv.x + xv.y * xv.y + xv.z * xv.z + xv.w * xv.w;
  ss += __shfl_xor(ss, 1, 32);
  ss += __shfl_xor(ss, 2, 32);
  ss += __shfl_xor(ss, 4, 32);
  ss += __shfl_xor(ss, 8, 32);
  ss += __shfl_xor(ss, 16, 32);
  const float rs = rsqrtf(ss * (1.f / 128.f) + 1e-6f);

  float val[4] = {xv.x, xv.y, xv.z, xv.w};
#pragma unroll
  for (int u = 0; u < 4; ++u)
    val[u] = (float)(__bf16)(val[u] * rs) * scale[lane * 4 + u];  // bf16 round like reference

  const bool hi = lane >= 16;   // lanes 0-15 hold first half, 16-31 second half
#pragma unroll
  for (int u = 0; u < 4; ++u) {
    const float oth = __shfl_xor(val[u], 16, 32);
    const int e = lane * 4 + u;
    const int iidx = e & 63;
    const float inv = __expf(-(float)iidx * (9.210340371976184f / 64.f)); // 10000^(-i/64)
    const float ang = (float)pos * inv;
    float sn, cs;
    __sincosf(ang, &sn, &cs);
    const float o = hi ? (val[u] * cs + oth * sn) : (val[u] * cs - oth * sn);
    dst[e] = (__bf16)o;
  }
}

// ---------------- causal flash attention (WMMA + async K/V tile loads) ----------------
// grid: (T/64, NH, B), block: 128 (4 waves), wave w owns 16 q rows.
__global__ __launch_bounds__(128) void flash_kernel(const __bf16* __restrict__ qb,
                                                    const __bf16* __restrict__ kb,
                                                    const __bf16* __restrict__ vt,
                                                    __bf16* __restrict__ enc) {
  __shared__ __bf16 Ks[64][136];    // 64 s-rows x 128 h
  __shared__ __bf16 Vs[128][72];    // 128 h-rows x 64 s  (from transposed V)
  __shared__ __bf16 Ps[4][16][72];  // per-wave probs 16 x 64

  const int qblk = blockIdx.x;
  const int n    = blockIdx.y;
  const int b    = blockIdx.z;
  const int tid  = threadIdx.x;
  const int lane = tid & 31;
  const int wave = tid >> 5;
  const int kv   = n >> 1;          // GQA group size 2

  // Q fragments stay in registers (16 rows x 128 K = 4 A-frags)
  v16bf qfr[4];
  {
    const int trow = qblk * 64 + wave * 16 + (lane & 15);
    const __bf16* qp = qb + (size_t)(b * T_ + trow) * (NH_ * HD_) + n * HD_;
    const int klo = (lane >> 4) << 3;
#pragma unroll
    for (int ks = 0; ks < 4; ++ks) {
      FragU u;
      u.q[0] = *(const uint4*)(qp + ks * 32 + klo);
      u.q[1] = *(const uint4*)(qp + ks * 32 + klo + 16);
      qfr[ks] = u.v;
    }
  }

  v8f oacc[8];
#pragma unroll
  for (int i = 0; i < 8; ++i)
#pragma unroll
    for (int v = 0; v < 8; ++v) oacc[i][v] = 0.f;
  float mrow[8], lrow[8];
#pragma unroll
  for (int v = 0; v < 8; ++v) { mrow[v] = -3.0e38f; lrow[v] = 0.f; }

  const int rb = (lane >> 4) << 3;

  // per-thread async source/dest addressing for the K/V tiles
  const int krow = tid >> 1, kpart = tid & 1;
  const __bf16* ksrc0 = kb + (size_t)(b * T_ + krow) * (NKV_ * HD_) + kv * HD_ + kpart * 64;
  const unsigned kdst = lds_off(&Ks[krow][kpart * 64]);
  const __bf16* vsrc0 = vt + ((size_t)(b * NKV_ + kv) * HD_ + tid) * T_;
  const unsigned vdst = lds_off(&Vs[tid][0]);

  for (int sb = 0; sb <= qblk; ++sb) {
    const int s0 = sb * 64;
    __syncthreads();   // previous iteration's reads of Ks/Vs are done
    {
      const __bf16* ksp = ksrc0 + (size_t)s0 * (NKV_ * HD_);
      const __bf16* vsp = vsrc0 + s0;
      // 128B per thread each, contiguous on both global and LDS sides
      ASYNC_LD(kdst, ksp, 0);   ASYNC_LD(kdst, ksp, 16);
      ASYNC_LD(kdst, ksp, 32);  ASYNC_LD(kdst, ksp, 48);
      ASYNC_LD(kdst, ksp, 64);  ASYNC_LD(kdst, ksp, 80);
      ASYNC_LD(kdst, ksp, 96);  ASYNC_LD(kdst, ksp, 112);
      ASYNC_LD(vdst, vsp, 0);   ASYNC_LD(vdst, vsp, 16);
      ASYNC_LD(vdst, vsp, 32);  ASYNC_LD(vdst, vsp, 48);
      ASYNC_LD(vdst, vsp, 64);  ASYNC_LD(vdst, vsp, 80);
      ASYNC_LD(vdst, vsp, 96);  ASYNC_LD(vdst, vsp, 112);
    }
    wait_async0();
    __syncthreads();

    // S = Q K^T  (16 x 64 per wave)
    v8f sacc[4];
#pragma unroll
    for (int i = 0; i < 4; ++i)
#pragma unroll
      for (int v = 0; v < 8; ++v) sacc[i][v] = 0.f;
#pragma unroll
    for (int nt = 0; nt < 4; ++nt)
#pragma unroll
      for (int ks = 0; ks < 4; ++ks) {
        v16bf bfrag = load_frag_b(&Ks[nt * 16][ks * 32], 136, lane);
        sacc[nt] = wmma_bf16(qfr[ks], bfrag, sacc[nt]);
      }

    // scale -> bf16 round -> causal mask (reference semantics)
    float sval[4][8];
    const int tg0 = qblk * 64 + wave * 16 + rb;
#pragma unroll
    for (int nt = 0; nt < 4; ++nt) {
      const int sg = s0 + nt * 16 + (lane & 15);
#pragma unroll
      for (int v = 0; v < 8; ++v) {
        float f = (float)(__bf16)(sacc[nt][v] * SCALAR_);
        if (sg > tg0 + v) f = KMASK_;
        sval[nt][v] = f;
      }
    }

    // online softmax (row-wise; rows live in 16-lane half-wave groups)
#pragma unroll
    for (int v = 0; v < 8; ++v) {
      float mx = fmaxf(fmaxf(sval[0][v], sval[1][v]), fmaxf(sval[2][v], sval[3][v]));
      mx = fmaxf(mx, __shfl_xor(mx, 1, 32));
      mx = fmaxf(mx, __shfl_xor(mx, 2, 32));
      mx = fmaxf(mx, __shfl_xor(mx, 4, 32));
      mx = fmaxf(mx, __shfl_xor(mx, 8, 32));
      const float mnew  = fmaxf(mrow[v], mx);
      const float alpha = __expf(mrow[v] - mnew);
      mrow[v] = mnew;
      float psum = 0.f;
#pragma unroll
      for (int nt = 0; nt < 4; ++nt) {
        const float p = __expf(sval[nt][v] - mnew);
        psum += p;
        Ps[wave][rb + v][nt * 16 + (lane & 15)] = (__bf16)p;
      }
      psum += __shfl_xor(psum, 1, 32);
      psum += __shfl_xor(psum, 2, 32);
      psum += __shfl_xor(psum, 4, 32);
      psum += __shfl_xor(psum, 8, 32);
      lrow[v] = lrow[v] * alpha + psum;
#pragma unroll
      for (int i = 0; i < 8; ++i) oacc[i][v] *= alpha;
    }

    // O += P V   (K = 64 s, N = 128 h)
#pragma unroll
    for (int ks2 = 0; ks2 < 2; ++ks2) {
      v16bf pfrag = load_frag_a(&Ps[wave][0][ks2 * 32], 72, lane);
#pragma unroll
      for (int nt = 0; nt < 8; ++nt) {
        v16bf vfrag = load_frag_b(&Vs[nt * 16][ks2 * 32], 72, lane);
        oacc[nt] = wmma_bf16(pfrag, vfrag, oacc[nt]);
      }
    }
  }

  // epilogue: divide by row-sum, write encoded bf16 (token, n*128+h)
#pragma unroll
  for (int nt = 0; nt < 8; ++nt) {
    const int col = n * HD_ + nt * 16 + (lane & 15);
#pragma unroll
    for (int v = 0; v < 8; ++v) {
      const int t = qblk * 64 + wave * 16 + rb + v;
      const float val = oacc[nt][v] / lrow[v];
      enc[(size_t)(b * T_ + t) * (NH_ * HD_) + col] = (__bf16)val;
    }
  }
}

// ---------------- launch ----------------
extern "C" void kernel_launch(void* const* d_in, const int* in_sizes, int n_in,
                              void* d_out, int out_size, void* d_ws, size_t ws_size,
                              hipStream_t stream) {
  (void)in_sizes; (void)n_in; (void)out_size; (void)ws_size;
  const float* x       = (const float*)d_in[0];
  const int*   segpos  = (const int*)d_in[1];
  // d_in[2] attn_mask: causal, computed analytically in flash_kernel
  const float* q_w     = (const float*)d_in[3];
  const float* kv_w    = (const float*)d_in[4];
  const float* o_w     = (const float*)d_in[5];
  const float* q_scale = (const float*)d_in[6];
  const float* k_scale = (const float*)d_in[7];

  char* ws = (char*)d_ws;
  constexpr size_t MB = 1024 * 1024;
  __bf16* xb   = (__bf16*)(ws + 0);        // 16 MB  (4096 x 2048)
  __bf16* wq_t = (__bf16*)(ws + 16 * MB);  //  8 MB  (2048 x 2048)
  __bf16* wk_t = (__bf16*)(ws + 24 * MB);  //  4 MB  (1024 x 2048)
  __bf16* wv_t = (__bf16*)(ws + 28 * MB);  //  4 MB
  __bf16* wo_t = (__bf16*)(ws + 32 * MB);  //  8 MB  (2048 x 2048)
  float*  qf   = (float*) (ws + 40 * MB);  // 32 MB  (4096 x 2048 f32)
  float*  kf   = (float*) (ws + 72 * MB);  // 16 MB  (4096 x 1024 f32)
  __bf16* qb   = (__bf16*)(ws + 88 * MB);  // 16 MB
  __bf16* kb   = (__bf16*)(ws + 104 * MB); //  8 MB
  __bf16* vt   = (__bf16*)(ws + 112 * MB); //  8 MB  (B,KV,H,T)
  __bf16* enc  = (__bf16*)(ws + 120 * MB); // 16 MB

  pack_x_kernel  <<<(B_ * T_ * D_) / 256, 256, 0, stream>>>(x, xb);
  pack_wq_kernel <<<(2048 * 2048) / 256, 256, 0, stream>>>(q_w, wq_t);
  pack_wkv_kernel<<<(2 * 1024 * 2048) / 256, 256, 0, stream>>>(kv_w, wk_t, wv_t);
  pack_wo_kernel <<<(2048 * 2048) / 256, 256, 0, stream>>>(o_w, wo_t);

  gemm_kernel<0><<<dim3(2048 / 128, 4096 / 128), 256, 0, stream>>>(xb, wq_t, qf, 4096, 2048, 2048);
  gemm_kernel<0><<<dim3(1024 / 128, 4096 / 128), 256, 0, stream>>>(xb, wk_t, kf, 4096, 1024, 2048);
  gemm_kernel<2><<<dim3(1024 / 128, 4096 / 128), 256, 0, stream>>>(xb, wv_t, vt, 4096, 1024, 2048);

  normrope_kernel<<<B_ * T_ * 24, 32, 0, stream>>>(qf, kf, segpos, q_scale, k_scale, qb, kb);

  flash_kernel<<<dim3(T_ / 64, NH_, B_), 128, 0, stream>>>(qb, kb, vt, enc);

  gemm_kernel<0><<<dim3(2048 / 128, 4096 / 128), 256, 0, stream>>>(enc, wo_t, (float*)d_out, 4096, 2048, 2048);
}
